// Z100_MLA_28948079575447
// MI455X (gfx1250) — matrix operand order repaired and checked by reading, not compiled, not run on hardware
//
#include <hip/hip_runtime.h>

// ---------------------------------------------------------------------------
// MLA forward for MI455X (gfx1250, wave32, WMMA + TDM + async LDS copies).
// All heavy math on v_wmma_f32_16x16x32_bf16 (f32 accumulate).
// ---------------------------------------------------------------------------

#define DEV __device__ __forceinline__

typedef __attribute__((ext_vector_type(16))) __bf16 bf16x16;
typedef __attribute__((ext_vector_type(8)))  __bf16 bf16x8;
typedef __attribute__((ext_vector_type(8)))  float  f32x8;
typedef __attribute__((ext_vector_type(4)))  unsigned int u32x4;
typedef __attribute__((ext_vector_type(8)))  unsigned int u32x8;

constexpr int Bn   = 2;
constexpr int Sn   = 1024;
constexpr int DIMn = 4096;
constexpr int Hn   = 32;
constexpr int QLRn = 1536;
constexpr int KVLRn= 512;
constexpr int NOPEn= 128;
constexpr int QKDn = 192;   // NOPE + ROPE
constexpr int Vd   = 128;
constexpr int MROWS= Bn * Sn;             // 2048
constexpr float SCALEc = 0.07216878364870323f;  // 192^-0.5

DEV f32x8 wmma_bf16(bf16x16 a, bf16x16 b, f32x8 c) {
  // 8-arg form: (neg_a, A, neg_b, B, c_mod, C, reuse_a, reuse_b)
  return __builtin_amdgcn_wmma_f32_16x16x32_bf16(false, a, false, b, (short)0, c,
                                                 false, false);
}

// A-fragment (16x32 bf16): lane half selects contiguous K block:
// lanes 0-15: elems = K[k0..k0+15]; lanes 16-31: K[k0+16..k0+31].
DEV bf16x16 frag_a(const __bf16* row, int k0, int khalf) {
  const __bf16* p = row + k0 + 16 * khalf;
  bf16x8 lo = *(const bf16x8*)(p);
  bf16x8 hi = *(const bf16x8*)(p + 8);
  bf16x16 r;
#pragma unroll
  for (int i = 0; i < 8; ++i) { r[i] = lo[i]; r[i + 8] = hi[i]; }
  return r;
}

// B-fragment (32x16 bf16), B tile stored LDS-row-major as [N][ldk]:
// lane = N, contiguous 16 K values selected by lane half.
DEV bf16x16 frag_b(const __bf16* rowN, int khalf) {
  const __bf16* p = rowN + 16 * khalf;
  bf16x8 lo = *(const bf16x8*)(p);
  bf16x8 hi = *(const bf16x8*)(p + 8);
  bf16x16 r;
#pragma unroll
  for (int i = 0; i < 8; ++i) { r[i] = lo[i]; r[i + 8] = hi[i]; }
  return r;
}

// ---- CDNA5 data movers ----------------------------------------------------

// Async global->LDS 16B copy per lane (ASYNCcnt-tracked).
DEV void async_b128(unsigned lds_byte_off, const void* gaddr) {
  asm volatile("global_load_async_to_lds_b128 %0, %1, off"
               :: "v"(lds_byte_off), "v"(gaddr)
               : "memory");
}
DEV void wait_async0() {
  asm volatile("s_wait_asynccnt 0x0" ::: "memory");
}

// Tensor Data Mover: load a 2D bf16 tile [rows x 576] (row stride 576 elems)
// from global to LDS.  D# per CDNA5 ISA:
//  group0: count=1, lds_addr, global_addr[56:0], type=2
//  group1: data_size=1(2B), tensor_dim0=576, tensor_dim1=1024,
//          tile_dim0=576, tile_dim1=rows, tensor_dim0_stride=576
DEV void tdm_load_k_tile(unsigned lds_byte_off, const void* gptr, unsigned rows) {
  unsigned long long ga = (unsigned long long)(size_t)gptr;
  u32x4 g0;
  g0[0] = 1u;                                                // count=1 (user D#)
  g0[1] = lds_byte_off;                                      // lds_addr
  g0[2] = (unsigned)(ga & 0xFFFFFFFFu);                      // global_addr lo
  g0[3] = (unsigned)((ga >> 32) & 0x01FFFFFFu) | 0x80000000u; // addr hi, type=2
  u32x8 g1;
  g1[0] = 0x00010000u;          // workgroup_mask=0, data_size=1 (2 bytes)
  g1[1] = (576u & 0xFFFFu) << 16;   // tensor_dim0[15:0]
  g1[2] = (1024u & 0xFFFFu) << 16;  // tensor_dim0[31:16]=0 | tensor_dim1[15:0]
  g1[3] = 576u << 16;               // tensor_dim1[31:16]=0 | tile_dim0
  g1[4] = rows & 0xFFFFu;           // tile_dim1 | tile_dim2=0
  g1[5] = 576u;                     // tensor_dim0_stride[31:0]
  g1[6] = 0u;                       // stride0 hi | tensor_dim1_stride lo
  g1[7] = 0u;
  asm volatile("tensor_load_to_lds %0, %1" :: "s"(g0), "s"(g1) : "memory");
}

// ---------------------------------------------------------------------------
// Generic bf16 WMMA GEMM:  C = alpha * A @ op(B)
//   BT = true : B is [N,K] row-major (x @ W^T style)
//   BT = false: B is [K,N] row-major
// Batched over blockIdx.z with element strides strideA/B/C.
// Block tile 128(M) x 128(N), 8 waves as 4(M) x 2(N), wave tile 32x64.
// ---------------------------------------------------------------------------
template <typename TA, bool BT, typename TC>
__global__ __launch_bounds__(256) void gemm_wmma(
    const TA* __restrict__ Aall, int lda, long long strideA,
    const float* __restrict__ Ball, int ldb, long long strideB,
    TC* __restrict__ Call, int ldc, long long strideC,
    int M, int N, int K, float alpha) {
  __shared__ __bf16 As[128 * 32];
  __shared__ __bf16 Bs[128 * 32];

  const TA* A = Aall + (size_t)blockIdx.z * strideA;
  const float* Bp = Ball + (size_t)blockIdx.z * strideB;
  TC* C = Call + (size_t)blockIdx.z * strideC;

  const int tid = threadIdx.x;
  const int lane = tid & 31, wave = tid >> 5;
  const int lrow = lane & 15, khalf = lane >> 4;
  const int m0 = blockIdx.y * 128, n0 = blockIdx.x * 128;
  const int wm0 = (wave >> 1) * 32;   // 4 waves along M
  const int wn0 = (wave & 1) * 64;    // 2 waves along N

  f32x8 acc[2][4] = {};

  for (int k0 = 0; k0 < K; k0 += 32) {
    // ---- stage A tile (128 x 32) -> bf16 LDS ----
#pragma unroll
    for (int it = 0; it < 4; ++it) {
      int slot = it * 256 + tid;
      int r = slot >> 3, c = (slot & 7) * 4;
      const TA* ap = A + (size_t)(m0 + r) * lda + (k0 + c);
      if constexpr (sizeof(TA) == 4) {
        float4 v = *(const float4*)ap;
        As[r * 32 + c + 0] = (__bf16)v.x;
        As[r * 32 + c + 1] = (__bf16)v.y;
        As[r * 32 + c + 2] = (__bf16)v.z;
        As[r * 32 + c + 3] = (__bf16)v.w;
      } else {
        *(uint2*)&As[r * 32 + c] = *(const uint2*)ap;  // already bf16
      }
    }
    // ---- stage B tile (128 x 32, as [N][K]) -> bf16 LDS ----
#pragma unroll
    for (int it = 0; it < 4; ++it) {
      int slot = it * 256 + tid;
      int n = slot >> 3, c = (slot & 7) * 4;
      if constexpr (BT) {
        if (n0 + n < N) {
          float4 v = *(const float4*)(Bp + (size_t)(n0 + n) * ldb + k0 + c);
          Bs[n * 32 + c + 0] = (__bf16)v.x;
          Bs[n * 32 + c + 1] = (__bf16)v.y;
          Bs[n * 32 + c + 2] = (__bf16)v.z;
          Bs[n * 32 + c + 3] = (__bf16)v.w;
        } else {
          Bs[n * 32 + c + 0] = (__bf16)0.f;
          Bs[n * 32 + c + 1] = (__bf16)0.f;
          Bs[n * 32 + c + 2] = (__bf16)0.f;
          Bs[n * 32 + c + 3] = (__bf16)0.f;
        }
      } else {
#pragma unroll
        for (int j = 0; j < 4; ++j) {
          float v = (n0 + n < N) ? Bp[(size_t)(k0 + c + j) * ldb + (n0 + n)] : 0.f;
          Bs[n * 32 + c + j] = (__bf16)v;
        }
      }
    }
    if (k0 + 32 < K) {  // hint next A tile into cache (global_prefetch_b8)
      __builtin_prefetch(A + (size_t)(m0 + (tid & 127)) * lda + k0 + 32, 0, 1);
    }
    __syncthreads();

    bf16x16 afr[2], bfr[4];
#pragma unroll
    for (int i = 0; i < 2; ++i)
      afr[i] = frag_a(&As[(wm0 + i * 16 + lrow) * 32], 0, khalf);
#pragma unroll
    for (int j = 0; j < 4; ++j)
      bfr[j] = frag_b(&Bs[(wn0 + j * 16 + lrow) * 32], khalf);
#pragma unroll
    for (int i = 0; i < 2; ++i)
#pragma unroll
      for (int j = 0; j < 4; ++j)
        acc[i][j] = wmma_bf16(afr[i], bfr[j], acc[i][j]);
    __syncthreads();
  }

  // C layout: vgpr e -> row e + 8*khalf, col = lane&15
#pragma unroll
  for (int i = 0; i < 2; ++i)
#pragma unroll
    for (int j = 0; j < 4; ++j) {
      int col = n0 + wn0 + j * 16 + lrow;
      if (col < N) {
#pragma unroll
        for (int e = 0; e < 8; ++e) {
          int row = m0 + wm0 + i * 16 + e + 8 * khalf;
          C[(size_t)row * ldc + col] = (TC)(alpha * acc[i][j][e]);
        }
      }
    }
}

// ---------------------------------------------------------------------------
// Row-wise RMSNorm, in place.  One 256-thread block per row.
// ---------------------------------------------------------------------------
__global__ __launch_bounds__(256) void rmsnorm_rows(float* __restrict__ X,
                                                    const float* __restrict__ w,
                                                    int W) {
  int row = blockIdx.x, tid = threadIdx.x, lane = tid & 31, wave = tid >> 5;
  float* xr = X + (size_t)row * W;
  __shared__ float red[8];
  float ss = 0.f;
  for (int c = tid; c < W; c += 256) { float v = xr[c]; ss += v * v; }
  for (int o = 16; o; o >>= 1) ss += __shfl_xor(ss, o, 32);
  if (lane == 0) red[wave] = ss;
  __syncthreads();
  if (tid == 0) {
    float t = 0.f;
    for (int i = 0; i < 8; ++i) t += red[i];
    red[0] = rsqrtf(t / (float)W + 1e-6f);
  }
  __syncthreads();
  float inv = red[0];
  for (int c = tid; c < W; c += 256) xr[c] = xr[c] * inv * w[c];
}

// ---------------------------------------------------------------------------
// kv post: Kcat[row][0:512] = bf16(rmsnorm(kv[:512]) * w), Kcat[512:576] = rope(k_pe)
// ---------------------------------------------------------------------------
__global__ __launch_bounds__(256) void kv_post(const float* __restrict__ KV,
                                               const float* __restrict__ wn,
                                               const float* __restrict__ fcos,
                                               const float* __restrict__ fsin,
                                               __bf16* __restrict__ Kcat) {
  int row = blockIdx.x;            // b*S + s
  int s = row & (Sn - 1);
  int tid = threadIdx.x, lane = tid & 31, wave = tid >> 5;
  const float* kv = KV + (size_t)row * 576;
  __shared__ float red[8];
  float ss = 0.f;
  for (int c = tid; c < 512; c += 256) { float v = kv[c]; ss += v * v; }
  for (int o = 16; o; o >>= 1) ss += __shfl_xor(ss, o, 32);
  if (lane == 0) red[wave] = ss;
  __syncthreads();
  if (tid == 0) {
    float t = 0.f;
    for (int i = 0; i < 8; ++i) t += red[i];
    red[0] = rsqrtf(t / 512.f + 1e-6f);
  }
  __syncthreads();
  float inv = red[0];
  __bf16* dst = Kcat + (size_t)row * 576;
  for (int c = tid; c < 512; c += 256) dst[c] = (__bf16)(kv[c] * inv * wn[c]);
  if (tid < 32) {
    int j = tid;
    float x0 = kv[512 + 2 * j], x1 = kv[513 + 2 * j];
    float cc = fcos[s * 32 + j], sn = fsin[s * 32 + j];
    dst[512 + 2 * j] = (__bf16)(x0 * cc - x1 * sn);
    dst[513 + 2 * j] = (__bf16)(x0 * sn + x1 * cc);
  }
}

// ---------------------------------------------------------------------------
// rope + pack q_pe (pre-scaled by SCALE) into Qcat[..., 512:576]
// ---------------------------------------------------------------------------
__global__ __launch_bounds__(256) void q_rope_pack(const float* __restrict__ Q,
                                                   const float* __restrict__ fcos,
                                                   const float* __restrict__ fsin,
                                                   __bf16* __restrict__ Qcat) {
  int idx = blockIdx.x * 256 + threadIdx.x;   // MROWS*H*32 total, exact grid
  int j = idx & 31;
  int h = (idx >> 5) & 31;
  int row = idx >> 10;                        // b*S + s
  int s = row & (Sn - 1);
  const float* q = Q + (size_t)row * (Hn * QKDn) + h * QKDn + NOPEn;
  float x0 = q[2 * j], x1 = q[2 * j + 1];
  float cc = fcos[s * 32 + j], sn = fsin[s * 32 + j];
  __bf16* dst = Qcat + ((size_t)row * Hn + h) * 576 + 512;
  dst[2 * j]     = (__bf16)((x0 * cc - x1 * sn) * SCALEc);
  dst[2 * j + 1] = (__bf16)((x0 * sn + x1 * cc) * SCALEc);
}

// ---------------------------------------------------------------------------
// Fused MLA attention core. One block per (b, h, 16-row query tile).
//  - Q tile staged by async global->LDS (ASYNCcnt path)
//  - K tiles (128 x 576 bf16, 144KB) staged by ONE tensor_load_to_lds each
//    (TENSORcnt path); 18 WMMAs per wave per barrier pair in phase 1.
//  - f32 softmax with causal mask; P bf16.
//  - P @ kv_c reuses the TDM-staged row-major tile; B fragments gathered
//    transposed from LDS; 16 WMMAs per wave per barrier pair.
// LDS: sQ 18432 | sK 147456 | sS 65536 | sP 32768  = 264192 B (<=320KB/WG)
// ---------------------------------------------------------------------------
__global__ __launch_bounds__(256) void mla_attn(const __bf16* __restrict__ Qcat,
                                                const __bf16* __restrict__ Kcat,
                                                __bf16* __restrict__ Olat) {
  extern __shared__ char smem[];
  __bf16* sQ = (__bf16*)(smem);                    // 16*576*2    = 18432 B
  __bf16* sK = (__bf16*)(smem + 18432);            // 128*576*2   = 147456 B
  float*  sS = (float*) (smem + 165888);           // 16*1024*4   = 65536 B
  __bf16* sP = (__bf16*)(smem + 231424);           // 16*1024*2   = 32768 B

  const int tid = threadIdx.x, lane = tid & 31, wave = tid >> 5;
  const int lrow = lane & 15, khalf = lane >> 4;
  const int s0 = blockIdx.x * 16;
  const int h = blockIdx.y, b = blockIdx.z;

  const __bf16* Qb = Qcat + ((size_t)(b * Sn + s0) * Hn + h) * 576;
  const __bf16* Kc = Kcat + (size_t)b * Sn * 576;
  const unsigned sQ_off = (unsigned)(size_t)sQ;   // LDS byte offset (addr[31:0])
  const unsigned sK_off = (unsigned)(size_t)sK;

  // ---- phase 0: async-copy Q tile (16 x 576 bf16 = 1152 x 16B chunks) ----
  for (int idx = tid; idx < 1152; idx += 256) {
    int r = idx / 72, ch = idx - r * 72;
    async_b128(sQ_off + r * 1152 + ch * 16, Qb + (size_t)r * (Hn * 576) + ch * 8);
  }
  wait_async0();
  __syncthreads();

  const int s_last = s0 + 15;
  // ---- phase 1: scores (one TDM tile + 18 WMMAs/wave per 128 t-columns) ----
  for (int t0 = 0; t0 <= s_last; t0 += 128) {
    if (wave == 0) {
      tdm_load_k_tile(sK_off, Kc + (size_t)t0 * 576, 128);
      __builtin_amdgcn_s_wait_tensorcnt(0);
    }
    __syncthreads();
    f32x8 acc = {};
#pragma unroll
    for (int k0 = 0; k0 < 576; k0 += 32) {
      bf16x16 a  = frag_a(&sQ[lrow * 576], k0, khalf);
      bf16x16 bb = frag_b(&sK[(wave * 16 + lrow) * 576 + k0], khalf);
      acc = wmma_bf16(a, bb, acc);
    }
#pragma unroll
    for (int e = 0; e < 8; ++e)
      sS[(e + 8 * khalf) * 1024 + t0 + wave * 16 + lrow] = acc[e];
    __syncthreads();   // protect sK before the next TDM overwrite
  }

  int tpad = ((s0 + 16 + 31) >> 5) << 5;
  if (tpad > Sn) tpad = Sn;

  // ---- phase 2: causal softmax, 2 rows per wave ----
  for (int rr = 0; rr < 2; ++rr) {
    int r = wave * 2 + rr;
    int nv = s0 + r + 1;               // valid keys: t in [0, nv)
    float* srow = sS + r * 1024;
    float mx = -3.0e38f;
    for (int t = lane; t < nv; t += 32) mx = fmaxf(mx, srow[t]);
    for (int o = 16; o; o >>= 1) mx = fmaxf(mx, __shfl_xor(mx, o, 32));
    float sum = 0.f;
    for (int t = lane; t < nv; t += 32) {
      float e = __expf(srow[t] - mx);
      srow[t] = e;
      sum += e;
    }
    for (int o = 16; o; o >>= 1) sum += __shfl_xor(sum, o, 32);
    float inv = 1.f / sum;
    __bf16* prow = sP + r * 1024;
    for (int t = lane; t < tpad; t += 32)
      prow[t] = (t < nv) ? (__bf16)(srow[t] * inv) : (__bf16)0.f;
  }
  __syncthreads();

  // ---- phase 3: o_lat = P @ kv_c (TDM tile reused row-major; transposed
  //      B fragments gathered from LDS; each wave owns 64 of 512 cols) ----
  f32x8 oacc[4] = {};
  for (int t0 = 0; t0 < tpad; t0 += 128) {
    if (wave == 0) {
      tdm_load_k_tile(sK_off, Kc + (size_t)t0 * 576, 128);
      __builtin_amdgcn_s_wait_tensorcnt(0);
    }
    __syncthreads();
    int kmax = tpad - t0; if (kmax > 128) kmax = 128;
    for (int ks = 0; ks < kmax; ks += 32) {
      bf16x16 a = frag_a(&sP[lrow * 1024], t0 + ks, khalf);
#pragma unroll
      for (int j = 0; j < 4; ++j) {
        const __bf16* bbase = &sK[(ks + 16 * khalf) * 576 + wave * 64 + j * 16 + lrow];
        bf16x16 bb;
#pragma unroll
        for (int e2 = 0; e2 < 16; ++e2) bb[e2] = bbase[e2 * 576];
        oacc[j] = wmma_bf16(a, bb, oacc[j]);
      }
    }
    __syncthreads();
  }
#pragma unroll
  for (int j = 0; j < 4; ++j)
#pragma unroll
    for (int e = 0; e < 8; ++e) {
      int srow = s0 + e + 8 * khalf;
      Olat[(size_t)(b * Sn + srow) * (Hn * 512) + h * 512 + wave * 64 + j * 16 +
           lrow] = (__bf16)oacc[j][e];
    }
}

// ---------------------------------------------------------------------------
// Host orchestration
// ---------------------------------------------------------------------------
extern "C" void kernel_launch(void* const* d_in, const int* in_sizes, int n_in,
                              void* d_out, int out_size, void* d_ws, size_t ws_size,
                              hipStream_t stream) {
  (void)in_sizes; (void)n_in; (void)out_size; (void)ws_size;
  const float* x       = (const float*)d_in[0];
  const float* wq_a    = (const float*)d_in[1];
  const float* q_norm  = (const float*)d_in[2];
  const float* wq_b    = (const float*)d_in[3];
  const float* wkv_a   = (const float*)d_in[4];
  const float* kv_norm = (const float*)d_in[5];
  const float* wkv_b   = (const float*)d_in[6];
  const float* wo      = (const float*)d_in[7];
  const float* fcos    = (const float*)d_in[8];
  const float* fsin    = (const float*)d_in[9];
  // d_in[10] mask: exactly causal, regenerated analytically in-kernel.
  // d_in[11] start_pos == 0.
  float* out = (float*)d_out;

  // workspace layout (bytes, 256-aligned)
  constexpr size_t off_qpre = 0;                                            // 2048*1536*4
  constexpr size_t off_q    = off_qpre + (size_t)MROWS * QLRn * 4;          // 2048*6144*4
  constexpr size_t off_kv   = off_q + (size_t)MROWS * Hn * QKDn * 4;        // 2048*576*4
  constexpr size_t off_kcat = off_kv + (size_t)MROWS * 576 * 4;             // 2048*576*2
  constexpr size_t off_qcat = off_kcat + (size_t)MROWS * 576 * 2;           // 2048*32*576*2
  constexpr size_t off_olat = off_qcat + (size_t)MROWS * Hn * 576 * 2;      // 2048*16384*2

  char* w = (char*)d_ws;
  float*  q_pre  = (float*)(w + off_qpre);
  float*  qbuf   = (float*)(w + off_q);
  float*  kvbuf  = (float*)(w + off_kv);
  __bf16* Kcat   = (__bf16*)(w + off_kcat);
  __bf16* Qcat   = (__bf16*)(w + off_qcat);
  __bf16* Olat   = (__bf16*)(w + off_olat);
  float*  o_flat = qbuf;   // reuse q buffer after q is consumed (step 7)

  dim3 blk(256);

  // 1. q_pre = x @ wq_a^T            [2048,4096] x [1536,4096]^T
  gemm_wmma<float, true, float><<<dim3(QLRn / 128, MROWS / 128, 1), blk, 0, stream>>>(
      x, DIMn, 0, wq_a, DIMn, 0, q_pre, QLRn, 0, MROWS, QLRn, DIMn, 1.0f);
  // 2. q_lat = rmsnorm(q_pre) (in place)
  rmsnorm_rows<<<dim3(MROWS), blk, 0, stream>>>(q_pre, q_norm, QLRn);
  // 3. q = q_lat @ wq_b^T            [2048,1536] x [6144,1536]^T
  gemm_wmma<float, true, float><<<dim3(Hn * QKDn / 128, MROWS / 128, 1), blk, 0, stream>>>(
      q_pre, QLRn, 0, wq_b, QLRn, 0, qbuf, Hn * QKDn, 0, MROWS, Hn * QKDn, QLRn, 1.0f);
  // 4. kv = x @ wkv_a^T              [2048,4096] x [576,4096]^T  (N=576: edge-guarded)
  gemm_wmma<float, true, float><<<dim3(5, MROWS / 128, 1), blk, 0, stream>>>(
      x, DIMn, 0, wkv_a, DIMn, 0, kvbuf, 576, 0, MROWS, 576, DIMn, 1.0f);
  // 5. Kcat = [rmsnorm(kv_c)*w | rope(k_pe)] as bf16
  kv_post<<<dim3(MROWS), blk, 0, stream>>>(kvbuf, kv_norm, fcos, fsin, Kcat);
  // 6. ql_nope = q_nope @ W_UK  (batched per head), scaled by SCALE, bf16 -> Qcat[:, :512]
  gemm_wmma<float, false, __bf16><<<dim3(KVLRn / 128, MROWS / 128, Hn), blk, 0, stream>>>(
      qbuf, Hn * QKDn, (long long)QKDn,
      wkv_b, KVLRn, (long long)(NOPEn + Vd) * KVLRn,
      Qcat, Hn * 576, 576,
      MROWS, KVLRn, NOPEn, SCALEc);
  // 7. Qcat[:, 512:576] = rope(q_pe) * SCALE
  q_rope_pack<<<dim3((MROWS * Hn * 32) / 256), blk, 0, stream>>>(qbuf, fcos, fsin, Qcat);
  // 8. fused attention -> Olat bf16 [2048, H*512]
  hipFuncSetAttribute((const void*)mla_attn,
                      hipFuncAttributeMaxDynamicSharedMemorySize, 264192);
  mla_attn<<<dim3(Sn / 16, Hn, Bn), blk, 264192, stream>>>(Qcat, Kcat, Olat);
  // 9. o = o_lat @ W_UV^T (batched per head) -> o_flat [2048, 4096]
  gemm_wmma<__bf16, true, float><<<dim3(1, MROWS / 128, Hn), blk, 0, stream>>>(
      Olat, Hn * KVLRn, (long long)KVLRn,
      wkv_b + (size_t)NOPEn * KVLRn, KVLRn, (long long)(NOPEn + Vd) * KVLRn,
      o_flat, Hn * Vd, (long long)Vd,
      MROWS, Vd, KVLRn, 1.0f);
  // 10. out = o_flat @ wo^T          [2048,4096] x [4096,4096]^T
  gemm_wmma<float, true, float><<<dim3(DIMn / 128, MROWS / 128, 1), blk, 0, stream>>>(
      o_flat, Hn * Vd, 0, wo, Hn * Vd, 0, out, DIMn, 0, MROWS, DIMn, Hn * Vd, 1.0f);
}
